// MultiHA_54554674593797
// MI455X (gfx1250) — compile-verified, matrix operand order checked by reading, not run on hardware
//
#include <hip/hip_runtime.h>

#define DEV __device__ __forceinline__

typedef __attribute__((ext_vector_type(16))) __bf16 v16bf;
typedef __attribute__((ext_vector_type(8)))  __bf16 v8bf;
typedef __attribute__((ext_vector_type(8)))  float  v8f;

constexpr int Dm  = 1024;   // model dim
constexpr int Sm  = 2048;   // sequence length
constexpr int Bm  = 2;      // batch
constexpr int NHm = 16;     // heads
constexpr int HSm = 64;     // head size
constexpr int Mrows = Bm * Sm;  // 4096 token rows

// ---------------------------------------------------------------------------
// Fragment loader for v_wmma_f32_16x16x32_bf16 (global memory).
// A (16x32, MxK) lane layout: lane = m + 16*g; elems 0..7 -> K = 8g+j,
// elems 8..15 -> K = 16+8g+j.  B mirrors it with lane&15 selecting N.
// ---------------------------------------------------------------------------
DEV v16bf load_frag(const __bf16* row_base, int k0, int lane) {
  int g = (lane >> 4) & 1;
  const __bf16* p = row_base + k0 + 8 * g;
  v8bf lo = *(const v8bf*)(p);
  v8bf hi = *(const v8bf*)(p + 16);
  v16bf f;
#pragma unroll
  for (int j = 0; j < 8; ++j) { f[j] = lo[j]; f[8 + j] = hi[j]; }
  return f;
}

// Same fragment shape, but sourced from an LDS row (32 bf16 per row).
DEV v16bf lds_frag(const __bf16* row_base, int lane) {
  int g = (lane >> 4) & 1;
  const __bf16* p = row_base + 8 * g;
  v8bf lo = *(const v8bf*)(p);
  v8bf hi = *(const v8bf*)(p + 16);
  v16bf f;
#pragma unroll
  for (int j = 0; j < 8; ++j) { f[j] = lo[j]; f[8 + j] = hi[j]; }
  return f;
}

DEV v8f wmma_bf16(v16bf a, v16bf b, v8f c) {
  return __builtin_amdgcn_wmma_f32_16x16x32_bf16(false, a, false, b,
                                                 (short)0, c, false, false);
}

// ---------------------------------------------------------------------------
// CDNA5 async global->LDS copy (16B per lane, one instruction per wave;
// tracked by ASYNCcnt).  dst is the LDS byte offset in a VGPR.
// ---------------------------------------------------------------------------
DEV void async_copy_b128(__bf16* lds_dst, const __bf16* gsrc) {
  unsigned int dst = (unsigned int)(unsigned long long)(void*)lds_dst;
  asm volatile("global_load_async_to_lds_b128 %0, %1, off"
               :: "v"(dst), "v"(gsrc) : "memory");
}

// ---------------------------------------------------------------------------
// f32 -> bf16 conversion
// ---------------------------------------------------------------------------
__global__ void mha_cvt_bf16(const float* __restrict__ in,
                             __bf16* __restrict__ out, int n) {
  int i = blockIdx.x * blockDim.x + threadIdx.x;
  if (i < n) out[i] = (__bf16)in[i];
}

// ---------------------------------------------------------------------------
// Block-cooperative GEMM core: block tile 128(M) x 64(N), K = 1024.
// All 8 waves share one 64x32 B tile per k-step, staged into LDS with
// async loads.  Statically double-buffered (k-loop unrolled by 2); a
// sched_group_barrier pins the compute region to "8 DS reads, then 4
// matrix ops" so the ds_load_b128 latencies overlap instead of serializing.
// ---------------------------------------------------------------------------
DEV void stage_issue(const __bf16* __restrict__ W, int n0, int k0,
                     __bf16* BsStage, int tid) {
  int i = tid >> 2;                    // 0..63: B row (output feature)
  int c = tid & 3;                     // 16B chunk within the 64B row
  const __bf16* src = W + (size_t)(n0 + i) * Dm + k0 + c * 8;
  async_copy_b128(BsStage + i * 32 + c * 8, src);
}

DEV void compute_step(const __bf16* stage, v16bf a, int ln, int lane,
                      v8f acc[4]) {
  v16bf b0 = lds_frag(stage + (0 * 16 + ln) * 32, lane);
  v16bf b1 = lds_frag(stage + (1 * 16 + ln) * 32, lane);
  v16bf b2 = lds_frag(stage + (2 * 16 + ln) * 32, lane);
  v16bf b3 = lds_frag(stage + (3 * 16 + ln) * 32, lane);
  acc[0] = wmma_bf16(a, b0, acc[0]);
  acc[1] = wmma_bf16(a, b1, acc[1]);
  acc[2] = wmma_bf16(a, b2, acc[2]);
  acc[3] = wmma_bf16(a, b3, acc[3]);
#if defined(__has_builtin)
#if __has_builtin(__builtin_amdgcn_sched_group_barrier)
  // Pipeline shape for this scheduling region: all 8 DS reads first,
  // then the 4 WMMAs (progressive dscnt waits instead of 4 full stalls).
  __builtin_amdgcn_sched_group_barrier(0x100, 8, 0);  // DS read group
  __builtin_amdgcn_sched_group_barrier(0x008, 4, 0);  // MFMA/WMMA group
#endif
#endif
}

DEV void gemm_block_128x64(const __bf16* __restrict__ A,
                           const __bf16* __restrict__ W,
                           int m0_wave, int n0, int lane, int tid,
                           __bf16 (*Bs)[32],     // [2*64][32] staging
                           v8f acc[4]) {
  int ln = lane & 15;
  const __bf16* arow = A + (size_t)(m0_wave + ln) * Dm;
  constexpr int NSTEP = Dm / 32;       // 32 k-steps (even)

  v16bf a0 = load_frag(arow, 0, lane);
  v16bf a1;
  stage_issue(W, n0, 0, &Bs[0][0], tid);

  for (int ks = 0; ks < NSTEP; ks += 2) {
    // ---- even step: compute stage0 / a0, prefetch stage1 / a1 ----
    stage_issue(W, n0, (ks + 1) * 32, &Bs[64][0], tid);
    a1 = load_frag(arow, (ks + 1) * 32, lane);
    asm volatile("s_wait_asynccnt 0x1" ::: "memory");
    __syncthreads();
    compute_step(&Bs[0][0], a0, ln, lane, acc);
    __syncthreads();

    // ---- odd step: compute stage1 / a1, prefetch stage0 / a0 ----
    int kn = (ks + 2 < NSTEP) ? (ks + 2) * 32 : 0;  // clamped dummy on last
    stage_issue(W, n0, kn, &Bs[0][0], tid);
    a0 = load_frag(arow, kn, lane);
    asm volatile("s_wait_asynccnt 0x1" ::: "memory");
    __syncthreads();
    compute_step(&Bs[64][0], a1, ln, lane, acc);
    __syncthreads();
  }
  asm volatile("s_wait_asynccnt 0x0" ::: "memory");
}

// ---------------------------------------------------------------------------
// Fused QKV projection.  blockIdx.y: 0=Q(from y), 1=K(from x), 2=V(from x).
// Q,K stored [B,H,S,hs]; V stored transposed [B,H,hs,S].
// ---------------------------------------------------------------------------
__global__ void mha_qkv(const __bf16* __restrict__ xb,
                        const __bf16* __restrict__ yb,
                        const __bf16* __restrict__ Wqb,
                        const __bf16* __restrict__ Wkb,
                        const __bf16* __restrict__ Wvb,
                        __bf16* __restrict__ Qb,
                        __bf16* __restrict__ Kb,
                        __bf16* __restrict__ Vtb) {
  __shared__ __bf16 Bs[2 * 64][32];    // double-buffered B tile (8KB)

  int tid  = threadIdx.x;
  int lane = tid & 31;
  int w    = tid >> 5;
  int bm   = blockIdx.x >> 4;          // 0..31 (M blocks of 128)
  int bn   = blockIdx.x & 15;          // 0..15 (N strips of 64)
  int m0   = bm * 128 + w * 16;
  int n0   = bn * 64;
  int z    = blockIdx.y;

  const __bf16* A = (z == 0) ? yb : xb;
  const __bf16* W = (z == 0) ? Wqb : (z == 1) ? Wkb : Wvb;

  v8f acc[4] = {};
  gemm_block_128x64(A, W, m0, n0, lane, tid, Bs, acc);

  int ln = lane & 15, g = lane >> 4;
#pragma unroll
  for (int t = 0; t < 4; ++t) {
#pragma unroll
    for (int r = 0; r < 8; ++r) {
      int bs = m0 + r + 8 * g;         // token row
      int n  = n0 + t * 16 + ln;       // output feature
      int b  = bs >> 11;
      int s  = bs & (Sm - 1);
      int h  = n >> 6;
      int d  = n & (HSm - 1);
      __bf16 val = (__bf16)acc[t][r];
      size_t bh = (size_t)b * NHm + h;
      if (z == 0)      Qb [(bh * Sm + s) * HSm + d] = val;
      else if (z == 1) Kb [(bh * Sm + s) * HSm + d] = val;
      else             Vtb[(bh * HSm + d) * Sm + s] = val;
    }
  }
}

// ---------------------------------------------------------------------------
// Flash attention.  grid = (S/128, B*NH), block = 256 (8 waves).
// Each wave owns 16 query rows; kv processed in tiles of 32.  K fragments
// are software-pipelined one tile ahead (static parity, loop unrolled x2)
// and V fragments are issued before the softmax so its VALU stretch hides
// their latency before the P*V WMMAs.
// ---------------------------------------------------------------------------
__global__ void mha_attn(const __bf16* __restrict__ Q,
                         const __bf16* __restrict__ K,
                         const __bf16* __restrict__ Vt,
                         const int*    __restrict__ pad_mask,
                         __bf16* __restrict__ O) {
  __shared__ float pbuf[8][16][32];    // per-wave P staging for transpose

  int lane = threadIdx.x & 31;
  int w    = threadIdx.x >> 5;
  int ln   = lane & 15, g = lane >> 4;

  int bh = blockIdx.y;
  int b  = bh >> 4;
  int h  = bh & (NHm - 1);
  int q0 = blockIdx.x * 128 + w * 16;

  const __bf16* Qrow = Q + ((size_t)bh * Sm + (q0 + ln)) * HSm;
  const __bf16* Kbh  = K + (size_t)bh * Sm * HSm;
  const __bf16* Vbh  = Vt + (size_t)bh * HSm * Sm;
  const int*    pm   = pad_mask + (size_t)b * Sm;

  v16bf qa0 = load_frag(Qrow, 0, lane);
  v16bf qa1 = load_frag(Qrow, 32, lane);

  v8f acc[4] = {};
  float mrow[8], lrow[8];
#pragma unroll
  for (int r = 0; r < 8; ++r) { mrow[r] = -1e30f; lrow[r] = 0.0f; }

  float* myp = &pbuf[w][0][0];
  const float scale = 0.125f;          // HS^-0.5

  auto load_ktile = [&](v16bf (&kf)[4], int kv) {
    const __bf16* Krow0 = Kbh + (size_t)(kv + ln) * HSm;
    const __bf16* Krow1 = Kbh + (size_t)(kv + 16 + ln) * HSm;
    kf[0] = load_frag(Krow0, 0,  lane);
    kf[1] = load_frag(Krow0, 32, lane);
    kf[2] = load_frag(Krow1, 0,  lane);
    kf[3] = load_frag(Krow1, 32, lane);
  };

  auto attn_step = [&](v16bf (&kf)[4], v16bf (&kn)[4], int kv0) {
    // V fragments for this tile: issue early, consumed after softmax
    v16bf vf[4];
#pragma unroll
    for (int t = 0; t < 4; ++t)
      vf[t] = load_frag(Vbh + (size_t)(t * 16 + ln) * Sm, kv0, lane);
    // prefetch next K tile (clamped dummy on the final tile)
    load_ktile(kn, (kv0 + 32 < Sm) ? kv0 + 32 : 0);

    v8f s0 = {}, s1 = {};
    s0 = wmma_bf16(qa0, kf[0], s0);
    s0 = wmma_bf16(qa1, kf[1], s0);
    s1 = wmma_bf16(qa0, kf[2], s1);
    s1 = wmma_bf16(qa1, kf[3], s1);

    bool ok0 = pm[kv0 + ln]      != 0;
    bool ok1 = pm[kv0 + 16 + ln] != 0;

    float ts0[8], ts1[8], tmax[8];
#pragma unroll
    for (int r = 0; r < 8; ++r) {
      ts0[r] = ok0 ? s0[r] * scale : -1e30f;
      ts1[r] = ok1 ? s1[r] * scale : -1e30f;
      tmax[r] = fmaxf(ts0[r], ts1[r]);
    }
#pragma unroll
    for (int off = 1; off < 16; off <<= 1)
#pragma unroll
      for (int r = 0; r < 8; ++r)
        tmax[r] = fmaxf(tmax[r], __shfl_xor(tmax[r], off, 32));

    float p0[8], p1[8], rsum[8];
#pragma unroll
    for (int r = 0; r < 8; ++r) {
      float mnew  = fmaxf(mrow[r], tmax[r]);
      float alpha = __expf(mrow[r] - mnew);
      p0[r] = ok0 ? __expf(ts0[r] - mnew) : 0.0f;
      p1[r] = ok1 ? __expf(ts1[r] - mnew) : 0.0f;
      rsum[r] = p0[r] + p1[r];
      mrow[r] = mnew;
      lrow[r] *= alpha;
#pragma unroll
      for (int t = 0; t < 4; ++t) acc[t][r] *= alpha;
    }
#pragma unroll
    for (int off = 1; off < 16; off <<= 1)
#pragma unroll
      for (int r = 0; r < 8; ++r)
        rsum[r] += __shfl_xor(rsum[r], off, 32);
#pragma unroll
    for (int r = 0; r < 8; ++r) lrow[r] += rsum[r];

    // transpose P (C/D layout -> A layout) through per-wave LDS
#pragma unroll
    for (int r = 0; r < 8; ++r) {
      myp[(r + 8 * g) * 32 + ln]      = p0[r];
      myp[(r + 8 * g) * 32 + 16 + ln] = p1[r];
    }
    __builtin_amdgcn_wave_barrier();
    asm volatile("s_wait_dscnt 0x0" ::: "memory");

    v16bf pa;
    const float* prow = myp + ln * 32;
#pragma unroll
    for (int j = 0; j < 8; ++j) {
      pa[j]     = (__bf16)prow[8 * g + j];
      pa[8 + j] = (__bf16)prow[16 + 8 * g + j];
    }
    __builtin_amdgcn_wave_barrier();

#pragma unroll
    for (int t = 0; t < 4; ++t)
      acc[t] = wmma_bf16(pa, vf[t], acc[t]);
  };

  v16bf kfA[4], kfB[4];
  load_ktile(kfA, 0);
  for (int kv0 = 0; kv0 < Sm; kv0 += 64) {
    attn_step(kfA, kfB, kv0);        // computes kv0,    prefetches kv0+32
    attn_step(kfB, kfA, kv0 + 32);   // computes kv0+32, prefetches kv0+64
  }

#pragma unroll
  for (int t = 0; t < 4; ++t) {
#pragma unroll
    for (int r = 0; r < 8; ++r) {
      int sq = q0 + r + 8 * g;
      int d  = h * HSm + t * 16 + ln;
      O[((size_t)b * Sm + sq) * Dm + d] = (__bf16)(acc[t][r] / lrow[r]);
    }
  }
}

// ---------------------------------------------------------------------------
// Output projection: out = O @ Wp^T + bp, f32 straight to d_out.
// ---------------------------------------------------------------------------
__global__ void mha_proj(const __bf16* __restrict__ Ob,
                         const __bf16* __restrict__ Wpb,
                         const float*  __restrict__ bp,
                         float* __restrict__ out) {
  __shared__ __bf16 Bs[2 * 64][32];

  int tid  = threadIdx.x;
  int lane = tid & 31;
  int w    = tid >> 5;
  int bm   = blockIdx.x >> 4;
  int bn   = blockIdx.x & 15;
  int m0   = bm * 128 + w * 16;
  int n0   = bn * 64;

  v8f acc[4] = {};
  gemm_block_128x64(Ob, Wpb, m0, n0, lane, tid, Bs, acc);

  int ln = lane & 15, g = lane >> 4;
#pragma unroll
  for (int t = 0; t < 4; ++t) {
    int n = n0 + t * 16 + ln;
    float bias = bp[n];
#pragma unroll
    for (int r = 0; r < 8; ++r) {
      int bs = m0 + r + 8 * g;
      out[(size_t)bs * Dm + n] = acc[t][r] + bias;
    }
  }
}

// ---------------------------------------------------------------------------
extern "C" void kernel_launch(void* const* d_in, const int* in_sizes, int n_in,
                              void* d_out, int out_size, void* d_ws, size_t ws_size,
                              hipStream_t stream) {
  const float* x  = (const float*)d_in[0];
  const float* y  = (const float*)d_in[1];
  const int*   pm = (const int*)  d_in[2];
  const float* Wq = (const float*)d_in[3];
  const float* Wk = (const float*)d_in[4];
  const float* Wv = (const float*)d_in[5];
  const float* Wp = (const float*)d_in[6];
  const float* bp = (const float*)d_in[7];
  float* out = (float*)d_out;

  const size_t NX = (size_t)Mrows * Dm;   // 4,194,304 activation elems
  const size_t NW = (size_t)Dm * Dm;      // 1,048,576 weight elems

  char* p = (char*)d_ws;
  auto take = [&](size_t bytes) { char* r = p; p += bytes; return r; };
  __bf16* xb  = (__bf16*)take(NX * 2);
  __bf16* yb  = (__bf16*)take(NX * 2);
  __bf16* Wqb = (__bf16*)take(NW * 2);
  __bf16* Wkb = (__bf16*)take(NW * 2);
  __bf16* Wvb = (__bf16*)take(NW * 2);
  __bf16* Wpb = (__bf16*)take(NW * 2);
  __bf16* Qb  = (__bf16*)take(NX * 2);
  __bf16* Kb  = (__bf16*)take(NX * 2);
  __bf16* Vtb = (__bf16*)take(NX * 2);
  __bf16* Ob  = (__bf16*)take(NX * 2);
  (void)ws_size; (void)in_sizes; (void)n_in; (void)out_size;

  const int T = 256;
  mha_cvt_bf16<<<(int)(NX / T), T, 0, stream>>>(x,  xb,  (int)NX);
  mha_cvt_bf16<<<(int)(NX / T), T, 0, stream>>>(y,  yb,  (int)NX);
  mha_cvt_bf16<<<(int)(NW / T), T, 0, stream>>>(Wq, Wqb, (int)NW);
  mha_cvt_bf16<<<(int)(NW / T), T, 0, stream>>>(Wk, Wkb, (int)NW);
  mha_cvt_bf16<<<(int)(NW / T), T, 0, stream>>>(Wv, Wvb, (int)NW);
  mha_cvt_bf16<<<(int)(NW / T), T, 0, stream>>>(Wp, Wpb, (int)NW);

  // 32 M-blocks x 16 N-strips = 512 blocks; z = {Q,K,V}
  mha_qkv<<<dim3(512, 3), 256, 0, stream>>>(xb, yb, Wqb, Wkb, Wvb, Qb, Kb, Vtb);

  // (S/128, B*NH)
  mha_attn<<<dim3(Sm / 128, Bm * NHm), 256, 0, stream>>>(Qb, Kb, Vtb, pm, Ob);

  mha_proj<<<512, 256, 0, stream>>>(Ob, Wpb, bp, out);
}
